// ARCANetNew_1443109012065
// MI455X (gfx1250) — compile-verified, matrix operand order checked by reading, not run on hardware
//
#include <hip/hip_runtime.h>
#include <math.h>

// ---------------------------------------------------------------------------
// Model constants (from the reference)
// ---------------------------------------------------------------------------
#define EMB      512
#define HEADS    2
#define NLAYERS  3
#define NPOOL    4
#define FCH      2048
#define NNODES   16384
#define NE0      65536
#define NEALL    (NE0 + NNODES)      // edges + self loops = 81920
#define BG       128
#define NCONC    32
#define GSLOPE   0.2f
#define LSLOPE   0.01f

#define ACT_NONE  0
#define ACT_RELU  1
#define ACT_LEAKY 2
#define ACT_SIG   3

typedef __attribute__((ext_vector_type(16))) _Float16 v16h;
typedef __attribute__((ext_vector_type(8)))  _Float16 v8h;
typedef __attribute__((ext_vector_type(8)))  float    v8f;

// ---------------------------------------------------------------------------
// Static device arena (float) + int arena. ~257 MB; avoids any runtime alloc
// and removes dependence on ws_size. Fully re-written each launch.
// ---------------------------------------------------------------------------
constexpr size_t S_H      = (size_t)NNODES * EMB;        // 8,388,608
constexpr size_t OFF_H    = 0;
constexpr size_t OFF_HIN  = OFF_H    + S_H;              // also h_cross in pooling
constexpr size_t OFF_XL   = OFF_HIN  + S_H;              // also t0 (leaky hsrc)
constexpr size_t OFF_XR   = OFF_XL   + S_H;              // also t (nn output)
constexpr size_t OFF_HOUT = OFF_XR   + S_H;
constexpr size_t OFF_BIG  = OFF_HOUT + S_H;              // 16384x1024 (covers 4096x2048)
constexpr size_t OFF_T2   = OFF_BIG  + 2 * S_H;          // 4096x512
constexpr size_t OFF_U    = OFF_T2   + (size_t)4096 * EMB;
constexpr size_t OFF_EA   = OFF_U    + (size_t)4096 * EMB;   // 81920x10
constexpr size_t OFF_LOOP = OFF_EA   + (size_t)NEALL * 10;   // 16384x10
constexpr size_t OFF_LOG  = OFF_LOOP + (size_t)NNODES * 10;  // 81920
constexpr size_t OFF_MAXB = OFF_LOG  + NEALL;                // 16384 (encoded u32)
constexpr size_t OFF_SUMB = OFF_MAXB + NNODES;
constexpr size_t OFF_DEG  = OFF_SUMB + NNODES;
constexpr size_t OFF_GS   = OFF_DEG  + NNODES;               // per-node gate scalar
constexpr size_t OFF_RG   = OFF_GS   + NNODES;               // 128x512
constexpr size_t OFF_GF   = OFF_RG   + (size_t)BG * EMB;
constexpr size_t OFF_S    = OFF_GF   + (size_t)BG * EMB;
constexpr size_t OFF_B    = OFF_S    + (size_t)BG * EMB;
constexpr size_t TOTAL_F  = OFF_B    + (size_t)BG * EMB;     // ~64.3M floats

__device__ __align__(256) float g_arena[TOTAL_F];
__device__ __align__(256) int   g_iarena[2 * NEALL];

// ---------------------------------------------------------------------------
// Helpers
// ---------------------------------------------------------------------------
__device__ __forceinline__ unsigned fenc(float f) {
  unsigned u = __float_as_uint(f);
  return (u & 0x80000000u) ? ~u : (u | 0x80000000u);   // order-preserving
}
__device__ __forceinline__ float fdec(unsigned u) {
  return __uint_as_float((u & 0x80000000u) ? (u & 0x7fffffffu) : ~u);
}
__device__ __forceinline__ float sigm(float x) { return 1.0f / (1.0f + expf(-x)); }
__device__ __forceinline__ float apply_act(float v, int act, float slope) {
  if (act == ACT_RELU)  return v > 0.f ? v : 0.f;
  if (act == ACT_LEAKY) return v > 0.f ? v : v * slope;
  if (act == ACT_SIG)   return sigm(v);
  return v;
}
__device__ __forceinline__ v16h cat16(v8h lo, v8h hi) {
  return __builtin_shufflevector(lo, hi, 0, 1, 2, 3, 4, 5, 6, 7,
                                 8, 9, 10, 11, 12, 13, 14, 15);
}

// ---------------------------------------------------------------------------
// WMMA GEMM:  C[M,N] = act(A[M,K] @ B[K,N] + bias)
// f32 in/out, f16 WMMA compute (V_WMMA_F32_16X16X32_F16), f32 accumulate.
// Block tile 128x128, 8 waves (2x4), wave tile 64x32 = 4x2 WMMA tiles, K-step 32.
// B is staged TRANSPOSED in LDS so both A and B fragments are 2x ds_load_b128.
// ---------------------------------------------------------------------------
__global__ __launch_bounds__(256) void gemm_wmma_k(
    const float* __restrict__ A, const float* __restrict__ B,
    const float* __restrict__ bias, float* __restrict__ C,
    int M, int N, int K, int lda, int ldb, int ldc, int act, float slope)
{
  __shared__ _Float16 sA [128][40];   // [m][k], row stride 80B (16B-aligned frags)
  __shared__ _Float16 sBt[128][40];   // [n][k] (transposed while staging)

  const int tid  = threadIdx.x;
  const int bm   = blockIdx.y * 128;
  const int bn   = blockIdx.x * 128;
  const int wave = tid >> 5, lane = tid & 31;
  const int wm   = wave >> 2, wn = wave & 3;      // 2 x 4 wave grid
  const int lr   = lane & 15;
  const int hi   = (lane >> 4) & 1;
  const int kh   = hi * 8;                        // frag K-half select

  // A staging: thread -> (row, 16 contiguous k)
  const int arow = tid >> 1;
  const int acol = (tid & 1) * 16;
  // B staging: thread -> (col, 16 k down a column)
  const int bcol = tid & 127;
  const int bk   = (tid >> 7) * 16;

  const bool a_row_ok = (bm + arow) < M;
  const bool b_col_ok = (bn + bcol) < N;
  const bool lda4     = ((lda & 3) == 0);

  v8f acc[4][2];
#pragma unroll
  for (int i = 0; i < 4; i++)
#pragma unroll
    for (int j = 0; j < 2; j++)
#pragma unroll
      for (int r = 0; r < 8; r++) acc[i][j][r] = 0.0f;

  for (int k0 = 0; k0 < K; k0 += 32) {
    __syncthreads();
    { // ---- stage A tile (128x32), f32 -> f16, vectorized fast path --------
      const float* ap = A + (size_t)(bm + arow) * lda + k0 + acol;
      v8h h0, h1;
      if (a_row_ok && (k0 + 32 <= K) && lda4) {
        __builtin_prefetch(ap + 32, 0, 1);
        const float4* p = (const float4*)ap;
        float4 f0 = p[0], f1 = p[1], f2 = p[2], f3 = p[3];
        h0[0] = (_Float16)f0.x; h0[1] = (_Float16)f0.y;
        h0[2] = (_Float16)f0.z; h0[3] = (_Float16)f0.w;
        h0[4] = (_Float16)f1.x; h0[5] = (_Float16)f1.y;
        h0[6] = (_Float16)f1.z; h0[7] = (_Float16)f1.w;
        h1[0] = (_Float16)f2.x; h1[1] = (_Float16)f2.y;
        h1[2] = (_Float16)f2.z; h1[3] = (_Float16)f2.w;
        h1[4] = (_Float16)f3.x; h1[5] = (_Float16)f3.y;
        h1[6] = (_Float16)f3.z; h1[7] = (_Float16)f3.w;
      } else {
#pragma unroll
        for (int c = 0; c < 16; c++) {
          float v = (a_row_ok && (k0 + acol + c) < K) ? ap[c] : 0.0f;
          if (c < 8) h0[c] = (_Float16)v; else h1[c - 8] = (_Float16)v;
        }
      }
      *(v8h*)&sA[arow][acol]     = h0;
      *(v8h*)&sA[arow][acol + 8] = h1;
    }
    { // ---- stage B tile (32x128) transposed: sBt[col][k] -------------------
      const float* bp = B + (size_t)(k0 + bk) * ldb + bn + bcol;
      v8h h0, h1;
      if (b_col_ok && (k0 + 32 <= K)) {
        __builtin_prefetch(bp + (size_t)32 * ldb, 0, 1);
#pragma unroll
        for (int j = 0; j < 8; j++) h0[j] = (_Float16)bp[(size_t)j * ldb];
#pragma unroll
        for (int j = 0; j < 8; j++) h1[j] = (_Float16)bp[(size_t)(j + 8) * ldb];
      } else {
#pragma unroll
        for (int j = 0; j < 16; j++) {
          float v = (b_col_ok && (k0 + bk + j) < K) ? bp[(size_t)j * ldb] : 0.0f;
          if (j < 8) h0[j] = (_Float16)v; else h1[j - 8] = (_Float16)v;
        }
      }
      *(v8h*)&sBt[bcol][bk]     = h0;
      *(v8h*)&sBt[bcol][bk + 8] = h1;
    }
    __syncthreads();

    // ---- fragments (ISA 7.12.2: lanes 0-15 K{0..7,16..23}, 16-31 K{8..15,24..31})
    v16h af[4], bf[2];
#pragma unroll
    for (int mi = 0; mi < 4; mi++) {
      int row = wm * 64 + mi * 16 + lr;
      v8h lo = *(const v8h*)&sA[row][kh];
      v8h hh = *(const v8h*)&sA[row][16 + kh];
      af[mi] = cat16(lo, hh);
    }
#pragma unroll
    for (int ni = 0; ni < 2; ni++) {
      int col = wn * 32 + ni * 16 + lr;
      v8h lo = *(const v8h*)&sBt[col][kh];
      v8h hh = *(const v8h*)&sBt[col][16 + kh];
      bf[ni] = cat16(lo, hh);
    }
#pragma unroll
    for (int mi = 0; mi < 4; mi++)
#pragma unroll
      for (int ni = 0; ni < 2; ni++)
        acc[mi][ni] = __builtin_amdgcn_wmma_f32_16x16x32_f16(
            false, af[mi], false, bf[ni], (short)0, acc[mi][ni], false, false);
  }

  // epilogue: C/D layout: lane<16 -> M=r, lane>=16 -> M=r+8; N = lr
#pragma unroll
  for (int mi = 0; mi < 4; mi++)
#pragma unroll
    for (int ni = 0; ni < 2; ni++)
#pragma unroll
      for (int r = 0; r < 8; r++) {
        int gm = bm + wm * 64 + mi * 16 + r + hi * 8;
        int gn = bn + wn * 32 + ni * 16 + lr;
        if (gm < M && gn < N) {
          float v = acc[mi][ni][r];
          if (bias) v += bias[gn];
          C[(size_t)gm * ldc + gn] = apply_act(v, act, slope);
        }
      }
}

// ---------------------------------------------------------------------------
// Utility / graph kernels
// ---------------------------------------------------------------------------
__global__ void fill_f_k(float* p, float v, int n) {
  int i = blockIdx.x * blockDim.x + threadIdx.x;
  if (i < n) p[i] = v;
}
__global__ void fill_enc_k(unsigned* p, int n) {
  int i = blockIdx.x * blockDim.x + threadIdx.x;
  if (i < n) p[i] = fenc(-INFINITY);
}
__global__ void edge_deg_k(const int* __restrict__ dst0, const float* __restrict__ ea0,
                           float* __restrict__ deg, float* __restrict__ loop, int E) {
  int e = blockIdx.x * blockDim.x + threadIdx.x;
  if (e >= E) return;
  int d = dst0[e];
  atomicAdd(&deg[d], 1.0f);
#pragma unroll
  for (int k = 0; k < 10; k++) atomicAdd(&loop[d * 10 + k], ea0[e * 10 + k]);
}
__global__ void build_edges_k(const int* __restrict__ ei, const float* __restrict__ ea0,
                              const float* __restrict__ loop, const float* __restrict__ deg,
                              int* __restrict__ srcA, int* __restrict__ dstA,
                              float* __restrict__ eaA) {
  int i = blockIdx.x * blockDim.x + threadIdx.x;
  if (i >= NEALL) return;
  if (i < NE0) {
    srcA[i] = ei[i];
    dstA[i] = ei[NE0 + i];
#pragma unroll
    for (int k = 0; k < 10; k++) eaA[(size_t)i * 10 + k] = ea0[(size_t)i * 10 + k];
  } else {
    int n = i - NE0;
    srcA[i] = n; dstA[i] = n;
    float inv = 1.0f / fmaxf(deg[n], 1.0f);
#pragma unroll
    for (int k = 0; k < 10; k++) eaA[(size_t)i * 10 + k] = loop[(size_t)n * 10 + k] * inv;
  }
}
// per-edge GATv2 logits; ea@We folded in registers (K=10)
__global__ void edge_logits_k(const float* __restrict__ xl, const float* __restrict__ xr,
                              const float* __restrict__ ea, const float* __restrict__ We,
                              int ldwe, const float* __restrict__ att,
                              const int* __restrict__ src, const int* __restrict__ dst,
                              float* __restrict__ logits, unsigned* __restrict__ maxb, int E) {
  int e = blockIdx.x * blockDim.x + threadIdx.x;
  if (e >= E) return;
  int s = src[e], d = dst[e];
  float ear[10];
#pragma unroll
  for (int k = 0; k < 10; k++) ear[k] = ea[(size_t)e * 10 + k];
  const float* pl = xl + (size_t)s * EMB;
  const float* pr = xr + (size_t)d * EMB;
  float acc = 0.0f;
  for (int c = 0; c < EMB; c++) {
    float em = 0.0f;
#pragma unroll
    for (int k = 0; k < 10; k++) em += ear[k] * We[k * ldwe + c];
    float m = pl[c] + pr[c] + em;
    m = m > 0.f ? m : m * GSLOPE;
    acc += m * att[c];
  }
  logits[e] = acc;
  atomicMax(&maxb[d], fenc(acc));
}
__global__ void seg_max_k(const float* __restrict__ vals, const int* __restrict__ seg,
                          unsigned* __restrict__ maxb, int n) {
  int i = blockIdx.x * blockDim.x + threadIdx.x;
  if (i < n) atomicMax(&maxb[seg[i]], fenc(vals[i]));
}
__global__ void seg_exp_k(float* __restrict__ vals, const int* __restrict__ seg,
                          const unsigned* __restrict__ maxb, float* __restrict__ sumb, int n) {
  int i = blockIdx.x * blockDim.x + threadIdx.x;
  if (i >= n) return;
  float v = expf(vals[i] - fdec(maxb[seg[i]]));
  vals[i] = v;
  atomicAdd(&sumb[seg[i]], v);
}
__global__ void edge_agg_k(float* __restrict__ hout, const float* __restrict__ xl,
                           const float* __restrict__ wexp, const float* __restrict__ sumb,
                           const int* __restrict__ src, const int* __restrict__ dst,
                           float scale) {
  int e = blockIdx.x;
  int s = src[e], d = dst[e];
  float alpha = scale * wexp[e] / sumb[d];
  for (int c = threadIdx.x; c < EMB; c += blockDim.x)
    atomicAdd(&hout[(size_t)d * EMB + c], alpha * xl[(size_t)s * EMB + c]);
}
__global__ void add_bias_k(float* __restrict__ h, const float* __restrict__ hout,
                           const float* __restrict__ bias, int n) {
  int i = blockIdx.x * blockDim.x + threadIdx.x;
  if (i < n) h[i] = hout[i] + bias[i % EMB];
}
__global__ void leaky_k(float* __restrict__ out, const float* __restrict__ in,
                        int n, float slope) {
  int i = blockIdx.x * blockDim.x + threadIdx.x;
  if (i >= n) return;
  float v = in[i];
  out[i] = v > 0.f ? v : v * slope;
}
__global__ void gate_res_k(float* __restrict__ h, const float* __restrict__ hout,
                           const float* __restrict__ bias, const float* __restrict__ gate,
                           int n) {
  int i = blockIdx.x * blockDim.x + threadIdx.x;
  if (i >= n) return;
  float g = sigm(gate[0]);
  h[i] = g * h[i] + (1.0f - g) * (hout[i] + bias[i % EMB]);
}
__global__ void hcross_k(float* __restrict__ out, const float* __restrict__ h,
                         const float* __restrict__ rg, const int* __restrict__ batch, int n) {
  int i = blockIdx.x * blockDim.x + threadIdx.x;
  if (i >= n) return;
  out[i] = h[i] + rg[(size_t)batch[i / EMB] * EMB + (i % EMB)];
}
__global__ void rowdot_k(const float* __restrict__ A, const float* __restrict__ v,
                         const float* __restrict__ b, float* __restrict__ out, int M, int K) {
  int m = blockIdx.x * blockDim.x + threadIdx.x;
  if (m >= M) return;
  const float* a = A + (size_t)m * K;
  float acc = 0.0f;
  for (int k = 0; k < K; k++) acc += a[k] * v[k];
  out[m] = acc + (b ? b[0] : 0.0f);
}
__global__ void pool_accum_k(const float* __restrict__ hsrc, const float* __restrict__ t,
                             const float* __restrict__ wexp, const float* __restrict__ sumb,
                             const int* __restrict__ batch, const float* __restrict__ gate,
                             float* __restrict__ gf) {
  int n = blockIdx.x;
  int bi = batch[n];
  float w = wexp[n] / sumb[bi];
  float a = sigm(gate[0]);
  for (int c = threadIdx.x; c < EMB; c += blockDim.x) {
    float v = a * hsrc[(size_t)n * EMB + c] + (1.0f - a) * t[(size_t)n * EMB + c];
    atomicAdd(&gf[(size_t)bi * EMB + c], w * v);
  }
}
__global__ void add_scaled_k(float* __restrict__ out, const float* __restrict__ in,
                             float s, int n) {
  int i = blockIdx.x * blockDim.x + threadIdx.x;
  if (i < n) out[i] += s * in[i];
}
__global__ void gate_combine_k(float* __restrict__ out, const float* __restrict__ x,
                               const float* __restrict__ t, const float* __restrict__ gate,
                               float scale, int n) {
  int i = blockIdx.x * blockDim.x + threadIdx.x;
  if (i >= n) return;
  float a = sigm(gate[0]);
  out[i] = (a * x[i] + (1.0f - a) * t[i]) * scale;
}
__global__ void head_u_k(const float* __restrict__ s, const float* __restrict__ b,
                         const float* __restrict__ concs, float* __restrict__ u, int n) {
  int i = blockIdx.x * blockDim.x + threadIdx.x;
  if (i >= n) return;
  int bi = i / (NCONC * EMB);
  int rem = i % (NCONC * EMB);
  int ci = rem / EMB, c = rem % EMB;
  float cc = fmaxf(concs[bi * NCONC + ci], 0.0f);
  float z = b[(size_t)bi * EMB + c] - s[(size_t)bi * EMB + c] * cc;
  u[i] = sigm(z);
}

// ---------------------------------------------------------------------------
// Host launcher
// ---------------------------------------------------------------------------
extern "C" void kernel_launch(void* const* d_in, const int* in_sizes, int n_in,
                              void* d_out, int out_size, void* d_ws, size_t ws_size,
                              hipStream_t stream) {
  (void)in_sizes; (void)n_in; (void)out_size; (void)d_ws; (void)ws_size;

  const float* expression = (const float*)d_in[0];
  const float* x          = (const float*)d_in[1];
  const int*   edge_index = (const int*)  d_in[2];
  const float* edge_attr  = (const float*)d_in[3];
  const int*   batch      = (const int*)  d_in[4];
  const float* concs      = (const float*)d_in[5];
  const float* gene_W     = (const float*)d_in[6];
  const float* gene_b     = (const float*)d_in[7];
  const float* g0_Wl      = (const float*)d_in[8];
  const float* g0_bl      = (const float*)d_in[9];
  const float* g0_Wr      = (const float*)d_in[10];
  const float* g0_br      = (const float*)d_in[11];
  const float* g0_We      = (const float*)d_in[12];
  const float* g0_att     = (const float*)d_in[13];
  const float* g0_bias    = (const float*)d_in[14];
  const float* g_Wl       = (const float*)d_in[15];
  const float* g_bl       = (const float*)d_in[16];
  const float* g_Wr       = (const float*)d_in[17];
  const float* g_br       = (const float*)d_in[18];
  const float* g_We       = (const float*)d_in[19];
  const float* g_att      = (const float*)d_in[20];
  const float* g_bias     = (const float*)d_in[21];
  const float* gat_gates  = (const float*)d_in[22];
  const float* p_gW1      = (const float*)d_in[23];
  const float* p_gb1      = (const float*)d_in[24];
  const float* p_gW2      = (const float*)d_in[25];
  const float* p_gb2      = (const float*)d_in[26];
  const float* p_nW1      = (const float*)d_in[27];
  const float* p_nb1      = (const float*)d_in[28];
  const float* p_nW2      = (const float*)d_in[29];
  const float* p_nb2      = (const float*)d_in[30];
  const float* p_gate     = (const float*)d_in[31];
  const float* fcs_W1     = (const float*)d_in[32];
  const float* fcs_b1     = (const float*)d_in[33];
  const float* fcs_W2     = (const float*)d_in[34];
  const float* fcs_b2     = (const float*)d_in[35];
  const float* fcs_gate   = (const float*)d_in[36];
  const float* fcb_W1     = (const float*)d_in[37];
  const float* fcb_b1     = (const float*)d_in[38];
  const float* fcb_W2     = (const float*)d_in[39];
  const float* fcb_b2     = (const float*)d_in[40];
  const float* fcb_gate   = (const float*)d_in[41];
  const float* fc_W1      = (const float*)d_in[42];
  const float* fc_b1      = (const float*)d_in[43];
  const float* fc_W2      = (const float*)d_in[44];
  const float* fc_b2      = (const float*)d_in[45];
  const float* fc_gate    = (const float*)d_in[46];
  const float* fc_Wo      = (const float*)d_in[47];
  const float* fc_bo      = (const float*)d_in[48];
  float* out = (float*)d_out;

  float* F = nullptr;
  int*   I = nullptr;
  hipGetSymbolAddress((void**)&F, HIP_SYMBOL(g_arena));
  hipGetSymbolAddress((void**)&I, HIP_SYMBOL(g_iarena));

  float*    h     = F + OFF_H;
  float*    hin   = F + OFF_HIN;    // also h_cross in pooling
  float*    xl    = F + OFF_XL;
  float*    xr    = F + OFF_XR;
  float*    hout  = F + OFF_HOUT;
  float*    big   = F + OFF_BIG;
  float*    t2    = F + OFF_T2;
  float*    u     = F + OFF_U;
  float*    eaA   = F + OFF_EA;
  float*    loopb = F + OFF_LOOP;
  float*    logit = F + OFF_LOG;
  unsigned* maxb  = (unsigned*)(F + OFF_MAXB);
  float*    sumb  = F + OFF_SUMB;
  float*    deg   = F + OFF_DEG;
  float*    gs    = F + OFF_GS;
  float*    rg    = F + OFF_RG;
  float*    gf    = F + OFF_GF;
  float*    svec  = F + OFF_S;
  float*    bvec  = F + OFF_B;
  int*      srcA  = I;
  int*      dstA  = I + NEALL;

  auto gemm = [&](const float* A, const float* B, const float* bias, float* C,
                  int M, int N, int K, int lda, int ldb, int ldc, int act, float slope) {
    dim3 grid((N + 127) / 128, (M + 127) / 128);
    gemm_wmma_k<<<grid, 256, 0, stream>>>(A, B, bias, C, M, N, K, lda, ldb, ldc, act, slope);
  };
  auto fillf = [&](float* p, float v, int n) {
    fill_f_k<<<(n + 255) / 256, 256, 0, stream>>>(p, v, n);
  };

  const int NC = NNODES * EMB;

  // ---- self loops: deg, mean edge_attr per dst, build augmented edge list --
  fillf(deg, 0.0f, NNODES);
  fillf(loopb, 0.0f, NNODES * 10);
  edge_deg_k<<<(NE0 + 255) / 256, 256, 0, stream>>>(edge_index + NE0, edge_attr, deg, loopb, NE0);
  build_edges_k<<<(NEALL + 255) / 256, 256, 0, stream>>>(edge_index, edge_attr, loopb, deg,
                                                         srcA, dstA, eaA);

  // ---- gene encoder: rg = relu(expression @ gene_W + gene_b) --------------
  gemm(expression, gene_W, gene_b, rg, BG, EMB, 2089, 2089, EMB, EMB, ACT_RELU, 0.f);

  // ---- GAT init layer (heads=1, concat) -----------------------------------
  gemm(x, g0_Wl, g0_bl, xl, NNODES, EMB, 79, 79, EMB, EMB, ACT_NONE, 0.f);
  gemm(x, g0_Wr, g0_br, xr, NNODES, EMB, 79, 79, EMB, EMB, ACT_NONE, 0.f);
  fill_enc_k<<<(NNODES + 255) / 256, 256, 0, stream>>>(maxb, NNODES);
  edge_logits_k<<<(NEALL + 255) / 256, 256, 0, stream>>>(xl, xr, eaA, g0_We, EMB, g0_att,
                                                         srcA, dstA, logit, maxb, NEALL);
  fillf(sumb, 0.0f, NNODES);
  seg_exp_k<<<(NEALL + 255) / 256, 256, 0, stream>>>(logit, dstA, maxb, sumb, NEALL);
  fillf(hout, 0.0f, NC);
  edge_agg_k<<<NEALL, 128, 0, stream>>>(hout, xl, logit, sumb, srcA, dstA, 1.0f);
  add_bias_k<<<(NC + 255) / 256, 256, 0, stream>>>(h, hout, g0_bias, NC);

  // ---- residual GAT layers (heads=2, mean) --------------------------------
  for (int i = 0; i < NLAYERS; i++) {
    leaky_k<<<(NC + 255) / 256, 256, 0, stream>>>(hin, h, NC, LSLOPE);
    fillf(hout, 0.0f, NC);
    for (int hh = 0; hh < HEADS; hh++) {
      const float* Wl = g_Wl + (size_t)i * EMB * (HEADS * EMB) + hh * EMB;
      const float* Wr = g_Wr + (size_t)i * EMB * (HEADS * EMB) + hh * EMB;
      const float* bl = g_bl + (size_t)i * (HEADS * EMB) + hh * EMB;
      const float* br = g_br + (size_t)i * (HEADS * EMB) + hh * EMB;
      const float* We = g_We + (size_t)i * 10 * (HEADS * EMB) + hh * EMB;
      const float* at = g_att + ((size_t)i * HEADS + hh) * EMB;
      gemm(hin, Wl, bl, xl, NNODES, EMB, EMB, EMB, HEADS * EMB, EMB, ACT_NONE, 0.f);
      gemm(hin, Wr, br, xr, NNODES, EMB, EMB, EMB, HEADS * EMB, EMB, ACT_NONE, 0.f);
      fill_enc_k<<<(NNODES + 255) / 256, 256, 0, stream>>>(maxb, NNODES);
      edge_logits_k<<<(NEALL + 255) / 256, 256, 0, stream>>>(xl, xr, eaA, We, HEADS * EMB, at,
                                                             srcA, dstA, logit, maxb, NEALL);
      fillf(sumb, 0.0f, NNODES);
      seg_exp_k<<<(NEALL + 255) / 256, 256, 0, stream>>>(logit, dstA, maxb, sumb, NEALL);
      edge_agg_k<<<NEALL, 128, 0, stream>>>(hout, xl, logit, sumb, srcA, dstA, 1.0f / HEADS);
    }
    gate_res_k<<<(NC + 255) / 256, 256, 0, stream>>>(h, hout, g_bias + (size_t)i * EMB,
                                                     gat_gates + i, NC);
  }

  // ---- pooling -------------------------------------------------------------
  hcross_k<<<(NC + 255) / 256, 256, 0, stream>>>(hin, h, rg, batch, NC);  // hin := h_cross
  fillf(gf, 0.0f, BG * EMB);
  for (int i = 0; i < NPOOL; i++) {
    for (int v = 0; v < 2; v++) {
      const float* hsrc = (v == 0) ? hin : h;
      // gate_nn: leaky(hsrc) -> leaky(@gW1+gb1) -> @gW2+gb2 (scalar per node)
      leaky_k<<<(NC + 255) / 256, 256, 0, stream>>>(xl, hsrc, NC, LSLOPE);
      gemm(xl, p_gW1 + (size_t)i * EMB * (2 * EMB), p_gb1 + (size_t)i * (2 * EMB), big,
           NNODES, 2 * EMB, EMB, EMB, 2 * EMB, 2 * EMB, ACT_LEAKY, LSLOPE);
      rowdot_k<<<(NNODES + 127) / 128, 128, 0, stream>>>(big, p_gW2 + (size_t)i * (2 * EMB),
                                                         p_gb2 + i, gs, NNODES, 2 * EMB);
      fill_enc_k<<<1, 256, 0, stream>>>(maxb, BG);
      fillf(sumb, 0.0f, BG);
      seg_max_k<<<(NNODES + 255) / 256, 256, 0, stream>>>(gs, batch, maxb, NNODES);
      seg_exp_k<<<(NNODES + 255) / 256, 256, 0, stream>>>(gs, batch, maxb, sumb, NNODES);
      // nn: t = leaky(hsrc@nW1+nb1)@nW2+nb2 ; gated with hsrc inside pool_accum
      gemm(hsrc, p_nW1 + (size_t)i * EMB * (2 * EMB), p_nb1 + (size_t)i * (2 * EMB), big,
           NNODES, 2 * EMB, EMB, EMB, 2 * EMB, 2 * EMB, ACT_LEAKY, LSLOPE);
      gemm(big, p_nW2 + (size_t)i * (2 * EMB) * EMB, p_nb2 + (size_t)i * EMB, xr,
           NNODES, EMB, 2 * EMB, 2 * EMB, EMB, EMB, ACT_NONE, 0.f);
      pool_accum_k<<<NNODES, 128, 0, stream>>>(hsrc, xr, gs, sumb, batch, p_gate + i, gf);
    }
  }
  add_scaled_k<<<(BG * EMB + 255) / 256, 256, 0, stream>>>(gf, rg, (float)NPOOL, BG * EMB);

  // ---- fc_s / fc_b gated residual MLPs + eval BatchNorm -------------------
  const float invbn = 1.0f / sqrtf(1.0f + 1e-5f);
  gemm(gf, fcs_W1, fcs_b1, big, BG, 2 * EMB, EMB, EMB, 2 * EMB, 2 * EMB, ACT_LEAKY, LSLOPE);
  gemm(big, fcs_W2, fcs_b2, t2, BG, EMB, 2 * EMB, 2 * EMB, EMB, EMB, ACT_NONE, 0.f);
  gate_combine_k<<<(BG * EMB + 255) / 256, 256, 0, stream>>>(svec, gf, t2, fcs_gate, invbn,
                                                             BG * EMB);
  gemm(gf, fcb_W1, fcb_b1, big, BG, 2 * EMB, EMB, EMB, 2 * EMB, 2 * EMB, ACT_LEAKY, LSLOPE);
  gemm(big, fcb_W2, fcb_b2, t2, BG, EMB, 2 * EMB, 2 * EMB, EMB, EMB, ACT_NONE, 0.f);
  gate_combine_k<<<(BG * EMB + 255) / 256, 256, 0, stream>>>(bvec, gf, t2, fcb_gate, invbn,
                                                             BG * EMB);

  // ---- concentration head -------------------------------------------------
  const int NU = BG * NCONC * EMB;  // 4096 x 512
  head_u_k<<<(NU + 255) / 256, 256, 0, stream>>>(svec, bvec, concs, u, NU);
  gemm(u, fc_W1, fc_b1, big, BG * NCONC, FCH, EMB, EMB, FCH, FCH, ACT_RELU, 0.f);
  gemm(big, fc_W2, fc_b2, t2, BG * NCONC, EMB, FCH, FCH, EMB, EMB, ACT_SIG, 0.f);
  gate_combine_k<<<(NU + 255) / 256, 256, 0, stream>>>(u, u, t2, fc_gate, 1.0f, NU);
  rowdot_k<<<(BG * NCONC + 127) / 128, 128, 0, stream>>>(u, fc_Wo, fc_bo, out, BG * NCONC, EMB);
}
